// MultiHeadAttention_82454782148911
// MI455X (gfx1250) — matrix operand order repaired and checked by reading, not compiled, not run on hardware
//
#include <hip/hip_runtime.h>
#include <hip/hip_bf16.h>

// Problem constants (match reference)
#define BB 2
#define SS 2048
#define DD 1024
#define HH 16
#define DHEAD 64
#define LN_EPS 1e-5f

typedef __bf16 bf16;
typedef __attribute__((ext_vector_type(8)))  bf16  bf16x8;
typedef __attribute__((ext_vector_type(16))) bf16  v16bf;
typedef __attribute__((ext_vector_type(8)))  float v8f;

// ---------------------------------------------------------------------------
// WMMA fragment helpers (gfx1250 wave32 layouts, CDNA5 ISA §7.12.2)
//
// A (16x32 bf16): lane l: row = l&15; elements 0..7 = K k0+hi*8 .. +7,
//                 elements 8..15 = K k0+hi*8+16 .. +23   (hi = l>>4)
// B (32x16 bf16, from BT row-major [N][K]): lane l: col N = n0+(l&15);
//                 elements 0..7 = K k0+hi*8 .. +7, elements 8..15 = K k0+hi*8+16 .. +23
// C/D (16x16 f32): element r: row M = r + hi*8, col N = l&15
// ---------------------------------------------------------------------------
__device__ __forceinline__ v16bf load_fragA(const bf16* base, int row, int ldk,
                                            int k0, int lane) {
  int hi = lane >> 4;
  const bf16* p = base + (size_t)row * ldk + k0 + hi * 8;
  union { v16bf v; bf16x8 h[2]; } u;
  u.h[0] = *(const bf16x8*)p;
  u.h[1] = *(const bf16x8*)(p + 16);
  return u.v;
}

__device__ __forceinline__ v16bf load_fragB(const bf16* bt, int n0, int ldk,
                                            int k0, int lane) {
  int l = lane & 15, hi = lane >> 4;
  const bf16* p = bt + (size_t)(n0 + l) * ldk + k0 + hi * 8;
  union { v16bf v; bf16x8 h[2]; } u;
  u.h[0] = *(const bf16x8*)p;
  u.h[1] = *(const bf16x8*)(p + 16);
  return u.v;
}

__device__ __forceinline__ v8f wmma_bf16(v16bf a, v16bf b, v8f c) {
  return __builtin_amdgcn_wmma_f32_16x16x32_bf16(false, a, false, b,
                                                 (short)0, c, false, false);
}

// ---------------------------------------------------------------------------
// Kernel 1: LayerNorm (fp32) -> bf16, one block per row, three inputs fused
// ---------------------------------------------------------------------------
__global__ void ln_cast_kernel(const float* __restrict__ q,
                               const float* __restrict__ k,
                               const float* __restrict__ v,
                               const float* __restrict__ gamma,
                               const float* __restrict__ beta,
                               bf16* __restrict__ lnq,
                               bf16* __restrict__ lnk,
                               bf16* __restrict__ lnv) {
  int rowid = blockIdx.x;                 // 0 .. 3*B*S-1
  int which = rowid / (BB * SS);
  int row   = rowid % (BB * SS);
  const float* src = (which == 0) ? q : (which == 1) ? k : v;
  bf16* dst        = (which == 0) ? lnq : (which == 1) ? lnk : lnv;
  src += (size_t)row * DD;
  dst += (size_t)row * DD;

  int t = threadIdx.x;                    // 256 threads, 4 elems each
  float x[4], s = 0.f, ss = 0.f;
#pragma unroll
  for (int i = 0; i < 4; ++i) {
    x[i] = src[t + 256 * i];
    s += x[i];
    ss += x[i] * x[i];
  }
  __shared__ float red[256];
  red[t] = s; __syncthreads();
  for (int off = 128; off > 0; off >>= 1) {
    if (t < off) red[t] += red[t + off];
    __syncthreads();
  }
  float mean = red[0] * (1.0f / DD);
  __syncthreads();
  red[t] = ss; __syncthreads();
  for (int off = 128; off > 0; off >>= 1) {
    if (t < off) red[t] += red[t + off];
    __syncthreads();
  }
  float var  = red[0] * (1.0f / DD) - mean * mean;
  float rstd = rsqrtf(var + LN_EPS);
#pragma unroll
  for (int i = 0; i < 4; ++i) {
    int c = t + 256 * i;
    dst[c] = (bf16)((x[i] - mean) * rstd * gamma[c] + beta[c]);
  }
}

// ---------------------------------------------------------------------------
// Kernel 2: transpose + cast the four 1024x1024 fp32 weights to bf16 [N][K]
// ---------------------------------------------------------------------------
__global__ void wtrans_kernel(const float* __restrict__ w0,
                              const float* __restrict__ w1,
                              const float* __restrict__ w2,
                              const float* __restrict__ w3,
                              bf16* __restrict__ t0, bf16* __restrict__ t1,
                              bf16* __restrict__ t2, bf16* __restrict__ t3) {
  int z = blockIdx.z;
  const float* W = (z == 0) ? w0 : (z == 1) ? w1 : (z == 2) ? w2 : w3;
  bf16*        T = (z == 0) ? t0 : (z == 1) ? t1 : (z == 2) ? t2 : t3;
  __shared__ float tile[32][33];
  int bx = blockIdx.x * 32;               // N
  int by = blockIdx.y * 32;               // K
  for (int i = threadIdx.y; i < 32; i += 8)
    tile[i][threadIdx.x] = W[(size_t)(by + i) * DD + bx + threadIdx.x];
  __syncthreads();
  for (int i = threadIdx.y; i < 32; i += 8)
    T[(size_t)(bx + i) * DD + by + threadIdx.x] = (bf16)tile[threadIdx.x][i];
}

// ---------------------------------------------------------------------------
// Kernel 3: WMMA GEMM  Y = A[4096xK=1024](bf16) * BT[N=1024][K](bf16) + bias
// mode 0/1: write bf16 head-major  [b][h][s][64]   (Q / K staging)
// mode 2:   write bf16 transposed  [b][h][64][s]   (V^T staging)
// mode 3:   write fp32 row-major   [b*S+s][1024]   (final out projection)
// Block = 128 threads = 4 waves arranged 2x2 -> 32x32 block tile.
// ---------------------------------------------------------------------------
__global__ void proj_kernel(const bf16* __restrict__ A,
                            const bf16* __restrict__ BT,
                            const float* __restrict__ bias,
                            void* __restrict__ out, int mode) {
  int lane = threadIdx.x & 31;
  int wid  = threadIdx.x >> 5;
  int m0 = blockIdx.x * 32 + (wid >> 1) * 16;
  int n0 = blockIdx.y * 32 + (wid & 1) * 16;
  int l = lane & 15, hi = lane >> 4;

  v8f acc = {0.f, 0.f, 0.f, 0.f, 0.f, 0.f, 0.f, 0.f};
  const bf16* arow = A + (size_t)(m0 + l) * DD;
#pragma unroll 4
  for (int k0 = 0; k0 < DD; k0 += 32) {
    if (k0 + 32 < DD) {                         // L2 prefetch hint
      __builtin_prefetch(arow + k0 + 32, 0, 1);
    }
    v16bf a = load_fragA(A, m0 + l, DD, k0, lane);
    v16bf b = load_fragB(BT, n0, DD, k0, lane);
    acc = wmma_bf16(a, b, acc);
  }

  int N = n0 + l;
  float bv = bias[N];
#pragma unroll
  for (int r = 0; r < 8; ++r) {
    int M = m0 + r + hi * 8;                    // global row in [0,4096)
    float y = acc[r] + bv;
    if (mode == 3) {
      ((float*)out)[(size_t)M * DD + N] = y;
    } else {
      int b_ = M / SS, s_ = M % SS;
      int h_ = N >> 6, d_ = N & 63;
      if (mode == 2) {                          // V^T: [b][h][d][s]
        ((bf16*)out)[(((size_t)(b_ * HH + h_) * DHEAD + d_)) * SS + s_] = (bf16)y;
      } else {                                  // Q/K: [b][h][s][d]
        ((bf16*)out)[(((size_t)(b_ * HH + h_) * SS + s_)) * DHEAD + d_] = (bf16)y;
      }
    }
  }
}

// ---------------------------------------------------------------------------
// Kernel 4: attention. One wave per 16-query tile of one (b,h).
// Pass 1: online row-max / row-sumexp over causal+pad key range (WMMA scores).
// Pass 2: recompute scores per pair of key tiles, write normalized probs to
//         d_out (fp32), stage prob tile in LDS as bf16, WMMA with V^T into
//         ctx accumulators; then vectorized zero-fill of the masked tail.
// ---------------------------------------------------------------------------
__global__ void attn_kernel(const bf16* __restrict__ Qb,
                            const bf16* __restrict__ Kb,
                            const bf16* __restrict__ Vt,
                            const int* __restrict__ lens,
                            float* __restrict__ attnOut,
                            bf16* __restrict__ ctx) {
  int lane = threadIdx.x & 31;
  int wid  = threadIdx.x >> 5;
  int qtg  = blockIdx.x * 4 + wid;        // 0 .. B*H*(S/16)-1
  int bh = qtg >> 7;                      // / (S/16)
  int qt = qtg & 127;
  int b_ = bh >> 4;
  int len = lens[b_];
  int q0 = qt * 16;
  int l = lane & 15, hi = lane >> 4;

  const bf16* Qh = Qb + (size_t)bh * SS * DHEAD;
  const bf16* Kh = Kb + (size_t)bh * SS * DHEAD;
  const bf16* Vh = Vt + (size_t)bh * DHEAD * SS;

  // Q fragments are invariant over the key loop: hoist (K-dim = 64 = 2 steps)
  v16bf qa0 = load_fragA(Qh, q0 + l, DHEAD, 0, lane);
  v16bf qa1 = load_fragA(Qh, q0 + l, DHEAD, 32, lane);

  int ntl = qt + 1;                       // causal bound in 16-key tiles
  int ptl = (len + 15) >> 4;              // pad bound
  int ntiles = (ntl < ptl) ? ntl : ptl;

  float m[8], sum[8];
#pragma unroll
  for (int r = 0; r < 8; ++r) { m[r] = -3.0e38f; sum[r] = 0.f; }

  // ---- pass 1: online max / sumexp (no HBM writes) ----
  for (int kt = 0; kt < ntiles; ++kt) {
    v8f sc = {0.f, 0.f, 0.f, 0.f, 0.f, 0.f, 0.f, 0.f};
    v16bf bk0 = load_fragB(Kh, kt * 16, DHEAD, 0, lane);
    v16bf bk1 = load_fragB(Kh, kt * 16, DHEAD, 32, lane);
    sc = wmma_bf16(qa0, bk0, sc);
    sc = wmma_bf16(qa1, bk1, sc);
    int ki = kt * 16 + l;
#pragma unroll
    for (int r = 0; r < 8; ++r) {
      int qrow = q0 + r + hi * 8;
      bool ok = (ki <= qrow) && (ki < len);
      float s = ok ? sc[r] * 0.125f : -3.0e38f;
      float tmax = s;
      for (int msk = 8; msk >= 1; msk >>= 1)
        tmax = fmaxf(tmax, __shfl_xor(tmax, msk, 32));
      float mnew = fmaxf(m[r], tmax);
      float p = ok ? __expf(s - mnew) : 0.f;
      float tsum = p;
      for (int msk = 8; msk >= 1; msk >>= 1)
        tsum += __shfl_xor(tsum, msk, 32);
      sum[r] = sum[r] * __expf(m[r] - mnew) + tsum;
      m[r] = mnew;
    }
  }

  float rinv[8];
#pragma unroll
  for (int r = 0; r < 8; ++r) rinv[r] = 1.0f / sum[r];   // sum >= 1 always

  // ---- pass 2: probabilities out + ctx accumulation ----
  __shared__ bf16 ptile_s[4][16][32];     // one 16x32 prob tile per wave
  bf16 (*pt)[32] = ptile_s[wid];
  float* arow = attnOut + ((size_t)bh * SS + q0) * SS;

  v8f cacc[4];
#pragma unroll
  for (int n = 0; n < 4; ++n)
    cacc[n] = (v8f){0.f, 0.f, 0.f, 0.f, 0.f, 0.f, 0.f, 0.f};

  int npair = (ntiles + 1) >> 1;
  for (int pp = 0; pp < npair; ++pp) {
#pragma unroll
    for (int sub = 0; sub < 2; ++sub) {
      int kt = pp * 2 + sub;
      if (kt < ntiles) {
        v8f sc = {0.f, 0.f, 0.f, 0.f, 0.f, 0.f, 0.f, 0.f};
        v16bf bk0 = load_fragB(Kh, kt * 16, DHEAD, 0, lane);
        v16bf bk1 = load_fragB(Kh, kt * 16, DHEAD, 32, lane);
        sc = wmma_bf16(qa0, bk0, sc);
        sc = wmma_bf16(qa1, bk1, sc);
        int ki = kt * 16 + l;
#pragma unroll
        for (int r = 0; r < 8; ++r) {
          int qrow = q0 + r + hi * 8;
          bool ok = (ki <= qrow) && (ki < len);
          float p = ok ? __expf(sc[r] * 0.125f - m[r]) * rinv[r] : 0.f;
          arow[(size_t)(r + hi * 8) * SS + kt * 16 + l] = p;   // fp32 probs
          pt[r + hi * 8][sub * 16 + l] = (bf16)p;
        }
      } else {
#pragma unroll
        for (int r = 0; r < 8; ++r)
          pt[r + hi * 8][sub * 16 + l] = (bf16)0.f;
      }
    }
    __builtin_amdgcn_wave_barrier();      // order LDS stores before loads

    // prob tile as K=32 A-fragment (in-order DS within a wave)
    union { v16bf v; bf16x8 h[2]; } ua;
    ua.h[0] = *(const bf16x8*)&pt[l][hi * 8];
    ua.h[1] = *(const bf16x8*)&pt[l][hi * 8 + 16];
    __builtin_amdgcn_wave_barrier();

    int kk = pp * 32;
#pragma unroll
    for (int n = 0; n < 4; ++n) {
      v16bf vb = load_fragB(Vh, n * 16, SS, kk, lane);
      cacc[n] = wmma_bf16(ua.v, vb, cacc[n]);
    }
  }

  // ctx -> [b][s][h*64+dv] as bf16 for the final projection
  int h_ = bh & 15;
#pragma unroll
  for (int n = 0; n < 4; ++n) {
#pragma unroll
    for (int r = 0; r < 8; ++r) {
      size_t row = (size_t)(b_ * SS + q0 + r + hi * 8);
      ctx[row * DD + h_ * DHEAD + n * 16 + l] = (bf16)cacc[n][r];
    }
  }

  // zero-fill the masked tail columns of the attention output (b128 stores)
  int ztail = ntiles * 16;
  float4 z4 = {0.f, 0.f, 0.f, 0.f};
#pragma unroll
  for (int r = 0; r < 8; ++r) {
    float* rowp = arow + (size_t)(r + hi * 8) * SS;
    for (int c = ztail + l * 4; c < SS; c += 64)
      *(float4*)(rowp + c) = z4;
  }
}

// ---------------------------------------------------------------------------
// Launch
// ---------------------------------------------------------------------------
extern "C" void kernel_launch(void* const* d_in, const int* in_sizes, int n_in,
                              void* d_out, int out_size, void* d_ws, size_t ws_size,
                              hipStream_t stream) {
  const float* q_seq = (const float*)d_in[0];
  const float* k_seq = (const float*)d_in[1];
  const float* v_seq = (const float*)d_in[2];
  const int*   lens  = (const int*)d_in[3];
  const float* gamma = (const float*)d_in[4];
  const float* beta  = (const float*)d_in[5];
  const float* Wq = (const float*)d_in[6];  const float* bq = (const float*)d_in[7];
  const float* Wk = (const float*)d_in[8];  const float* bk = (const float*)d_in[9];
  const float* Wv = (const float*)d_in[10]; const float* bv = (const float*)d_in[11];
  const float* Wo = (const float*)d_in[12]; const float* bo = (const float*)d_in[13];

  float* outp    = (float*)d_out;                       // [B,S,D]
  float* attnOut = outp + (size_t)BB * SS * DD;         // [B*H,S,S]

  char* w = (char*)d_ws;
  const size_t SEQ_BYTES = (size_t)BB * SS * DD * sizeof(bf16);   // 8 MB
  const size_t W_BYTES   = (size_t)DD * DD * sizeof(bf16);        // 2 MB
  bf16* lnQ = (bf16*)w; w += SEQ_BYTES;
  bf16* lnK = (bf16*)w; w += SEQ_BYTES;
  bf16* lnV = (bf16*)w; w += SEQ_BYTES;
  bf16* WqT = (bf16*)w; w += W_BYTES;
  bf16* WkT = (bf16*)w; w += W_BYTES;
  bf16* WvT = (bf16*)w; w += W_BYTES;
  bf16* WoT = (bf16*)w; w += W_BYTES;
  bf16* Qb  = (bf16*)w; w += SEQ_BYTES;                 // [b][h][s][64]
  bf16* Kb  = (bf16*)w; w += SEQ_BYTES;                 // [b][h][s][64]
  bf16* Vt  = (bf16*)w; w += SEQ_BYTES;                 // [b][h][64][s]
  bf16* ctx = (bf16*)w; w += SEQ_BYTES;                 // [b][s][h*64+dv]

  ln_cast_kernel<<<3 * BB * SS, 256, 0, stream>>>(q_seq, k_seq, v_seq,
                                                  gamma, beta, lnQ, lnK, lnV);

  wtrans_kernel<<<dim3(32, 32, 4), dim3(32, 8), 0, stream>>>(Wq, Wk, Wv, Wo,
                                                             WqT, WkT, WvT, WoT);

  dim3 pgrid(BB * SS / 32, DD / 32);                    // 128 x 32
  proj_kernel<<<pgrid, 128, 0, stream>>>(lnQ, WqT, bq, (void*)Qb, 0);
  proj_kernel<<<pgrid, 128, 0, stream>>>(lnK, WkT, bk, (void*)Kb, 1);
  proj_kernel<<<pgrid, 128, 0, stream>>>(lnV, WvT, bv, (void*)Vt, 2);

  attn_kernel<<<BB * HH * (SS / 16) / 4, 128, 0, stream>>>(Qb, Kb, Vt, lens,
                                                           attnOut, ctx);

  proj_kernel<<<pgrid, 128, 0, stream>>>(ctx, WoT, bo, (void*)outp, 3);
}
